// DecoderLayer_31086973288870
// MI455X (gfx1250) — compile-verified, matrix operand order checked by reading, not run on hardware
//
#include <hip/hip_runtime.h>
#include <cstddef>
#include <cstdint>

#define D_MODEL 2048
#define HEADS   16
#define HEAD_DIM 128
#define INNER   8192
#define SEQ     2048

typedef _Float16 v16h __attribute__((ext_vector_type(16)));
typedef _Float16 v4h  __attribute__((ext_vector_type(4)));
typedef _Float16 v2h  __attribute__((ext_vector_type(2)));
typedef float    v8f  __attribute__((ext_vector_type(8)));
typedef int      v4i  __attribute__((ext_vector_type(4)));

typedef __attribute__((address_space(1))) v4i av4i_g;   // global
typedef __attribute__((address_space(3))) v4i av4i_l;   // LDS

#if defined(__has_builtin)
#if __has_builtin(__builtin_amdgcn_global_load_async_to_lds_b128)
#define HAVE_ASYNC_LDS 1
#endif
#endif
#ifndef HAVE_ASYNC_LDS
#define HAVE_ASYNC_LDS 0
#endif

__device__ __forceinline__ v8f v8f_zero() {
    v8f z;
#pragma unroll
    for (int i = 0; i < 8; ++i) z[i] = 0.0f;
    return z;
}

// 16-byte global -> LDS copy. Uses the CDNA5 async-DMA path when available
// (tracked by ASYNCcnt, bypasses VGPRs); falls back to a plain generic copy.
__device__ __forceinline__ void async_copy16(const void* g, void* l) {
#if HAVE_ASYNC_LDS
    __builtin_amdgcn_global_load_async_to_lds_b128(
        (av4i_g*)(uintptr_t)g,
        (av4i_l*)(uint32_t)(uintptr_t)l,
        0, 0);
#else
    *(int4*)l = *(const int4*)g;
#endif
}

__device__ __forceinline__ void wait_async() {
#if HAVE_ASYNC_LDS
#if __has_builtin(__builtin_amdgcn_s_wait_asynccnt)
    __builtin_amdgcn_s_wait_asynccnt(0);
#else
    asm volatile("s_wait_asynccnt 0" ::: "memory");
#endif
#endif
}

// ---------------------------------------------------------------------------
// Weight prep: out[C,R] (f16) = transpose(in[R,C] f32)
// ---------------------------------------------------------------------------
__global__ __launch_bounds__(256) void transpose_f32_f16(
    const float* __restrict__ in, _Float16* __restrict__ out, int R, int C)
{
    __shared__ _Float16 t[32][33];
    const int tx = threadIdx.x, ty = threadIdx.y;     // block (32, 8)
    const int c0 = blockIdx.x * 32, r0 = blockIdx.y * 32;
#pragma unroll
    for (int i = 0; i < 4; ++i) {
        const int r = r0 + ty + i * 8;
        t[ty + i * 8][tx] = (_Float16)in[(size_t)r * C + c0 + tx];
    }
    __syncthreads();
#pragma unroll
    for (int i = 0; i < 4; ++i) {
        const int c = c0 + ty + i * 8;
        out[(size_t)c * R + r0 + tx] = t[tx][ty + i * 8];
    }
}

// Plain f32 -> f16 conversion (for x).
__global__ __launch_bounds__(256) void cvt_f32_f16(
    const float* __restrict__ in, _Float16* __restrict__ out, int n4)
{
    const int i = blockIdx.x * 256 + threadIdx.x;
    if (i < n4) {
        const float4 v = ((const float4*)in)[i];
        v4h h;
        h[0] = (_Float16)v.x; h[1] = (_Float16)v.y;
        h[2] = (_Float16)v.z; h[3] = (_Float16)v.w;
        ((v4h*)out)[i] = h;
    }
}

// ---------------------------------------------------------------------------
// GEMM: C = act(A[M,K] @ Bt[N,K]^T + bias), A/Bt f16 (Bt pre-transposed),
// accumulate f32. Block tile 128x128, K-step 32; 8 waves, each 4x2 WMMA tiles.
// Tiles staged global->LDS with async-DMA (no conversions in the hot loop).
// EPI: 0 bias, 1 bias+exact GELU.  O32/O16: which outputs to write.
// ---------------------------------------------------------------------------
#define BM 128
#define BN 128
#define BK 32
#define LDS_K 40   // f16 elems per row in LDS (16B-aligned stride: 80 bytes)

template <int EPI, bool O32, bool O16>
__global__ __launch_bounds__(256) void gemm_wmma_f16(
    const _Float16* __restrict__ A, const _Float16* __restrict__ Bt,
    const float* __restrict__ bias,
    float* __restrict__ C32, _Float16* __restrict__ C16,
    int N, int K)
{
    __shared__ _Float16 sA[BM * LDS_K];   // [row][k]
    __shared__ _Float16 sB[BN * LDS_K];   // [col][k]

    const int tid  = threadIdx.x;
    const int lane = tid & 31;
    const int wid  = tid >> 5;
    const int wm   = wid >> 2;           // 0..1
    const int wn   = wid & 3;            // 0..3
    const int bm   = blockIdx.y * BM;
    const int bn   = blockIdx.x * BN;
    const int l15  = lane & 15;
    const int kg   = (lane >> 4) << 3;   // 0 or 8

    v8f acc[4][2];
#pragma unroll
    for (int mi = 0; mi < 4; ++mi)
#pragma unroll
        for (int ni = 0; ni < 2; ++ni) acc[mi][ni] = v8f_zero();

    for (int k0 = 0; k0 < K; k0 += BK) {
        // ---- stage A and B tiles (each 128 rows x 32 f16 = 4 x 16B chunks) ----
#pragma unroll
        for (int p = 0; p < 2; ++p) {
            const int cidx = p * 256 + tid;            // 0..511
            const int r = cidx >> 2, part = (cidx & 3) * 8;
            async_copy16(A + (size_t)(bm + r) * K + k0 + part, &sA[r * LDS_K + part]);
        }
#pragma unroll
        for (int p = 0; p < 2; ++p) {
            const int cidx = p * 256 + tid;
            const int r = cidx >> 2, part = (cidx & 3) * 8;
            async_copy16(Bt + (size_t)(bn + r) * K + k0 + part, &sB[r * LDS_K + part]);
        }
        wait_async();
        __syncthreads();

        // ---- fragments (pairable 32-bit LDS reads) + WMMA ----
        v16h af[4], bfg[2];
#pragma unroll
        for (int mi = 0; mi < 4; ++mi) {
            const int row = wm * 64 + mi * 16 + l15;
            v2h* ap = (v2h*)&af[mi];
#pragma unroll
            for (int jp = 0; jp < 8; ++jp) {
                const int k = ((jp & 3) << 1) + kg + ((jp >> 2) << 4);
                ap[jp] = *(const v2h*)&sA[row * LDS_K + k];
            }
        }
#pragma unroll
        for (int ni = 0; ni < 2; ++ni) {
            const int col = wn * 32 + ni * 16 + l15;
            v2h* bp = (v2h*)&bfg[ni];
#pragma unroll
            for (int jp = 0; jp < 8; ++jp) {
                const int k = ((jp & 3) << 1) + kg + ((jp >> 2) << 4);
                bp[jp] = *(const v2h*)&sB[col * LDS_K + k];
            }
        }
#pragma unroll
        for (int mi = 0; mi < 4; ++mi)
#pragma unroll
            for (int ni = 0; ni < 2; ++ni)
                acc[mi][ni] = __builtin_amdgcn_wmma_f32_16x16x32_f16(
                    false, af[mi], false, bfg[ni], (short)0, acc[mi][ni], false, false);
        __syncthreads();
    }

    // ---- epilogue ----
#pragma unroll
    for (int mi = 0; mi < 4; ++mi) {
        const int row0 = bm + wm * 64 + mi * 16 + ((lane >> 4) << 3);
#pragma unroll
        for (int ni = 0; ni < 2; ++ni) {
            const int col = bn + wn * 32 + ni * 16 + l15;
            const float bv = bias[col];
#pragma unroll
            for (int i = 0; i < 8; ++i) {
                float v = acc[mi][ni][i] + bv;
                if (EPI == 1) v = 0.5f * v * (1.0f + erff(v * 0.70710678118654752f));
                if (O32) C32[(size_t)(row0 + i) * N + col] = v;
                if (O16) C16[(size_t)(row0 + i) * N + col] = (_Float16)v;
            }
        }
    }
}

// ---------------------------------------------------------------------------
// Causal flash attention over packed f16 QKV c[SEQ, 3*D_MODEL].
// grid = (HEADS, SEQ/64); 128 threads = 4 waves, 16 q rows per wave.
// KV tiles of 32: S = Q K^T (WMMA), online softmax, O += P V (WMMA).
// V tile staged once per block via async-DMA into LDS (shared by all waves).
// Output written f16 for the projection GEMM.
// ---------------------------------------------------------------------------
__global__ __launch_bounds__(128) void attn_flash(
    const _Float16* __restrict__ c, _Float16* __restrict__ out)
{
    __shared__ _Float16 sP[4][16 * 32];
    __shared__ _Float16 sV[32 * 136];          // [kv 0..31][dim 0..127], pad 8

    const int tid   = threadIdx.x;
    const int lane  = tid & 31;
    const int wid   = tid >> 5;                // 0..3
    const int head  = blockIdx.x;
    const int qblk  = blockIdx.y;              // 64 q rows per block
    const int qbase = qblk * 64 + wid * 16;
    const int l15   = lane & 15;
    const int half  = lane >> 4;
    const int kg    = half << 3;
    const size_t ldc = 3 * D_MODEL;
    const float scale = 0.022097086912079608f; // 1/sqrt(2048)

    // ---- preload Q fragments ----
    v16h qf[4];
    {
        const _Float16* qrow = c + (size_t)(qbase + l15) * ldc + head * HEAD_DIM;
#pragma unroll
        for (int kc = 0; kc < 4; ++kc) {
            v2h* qp = (v2h*)&qf[kc];
#pragma unroll
            for (int jp = 0; jp < 8; ++jp) {
                const int k = ((jp & 3) << 1) + kg + ((jp >> 2) << 4);
                qp[jp] = *(const v2h*)(qrow + kc * 32 + k);
            }
        }
    }

    v8f o[8];
#pragma unroll
    for (int vt = 0; vt < 8; ++vt) o[vt] = v8f_zero();
    float mrow[8], lrow[8];
#pragma unroll
    for (int i = 0; i < 8; ++i) { mrow[i] = -INFINITY; lrow[i] = 0.0f; }

    const int ntiles = qblk * 2 + 2;           // uniform across block
    for (int t = 0; t < ntiles; ++t) {
        const int kv0 = t * 32;

        // ---- kick off async V-tile staging (32 x 64 f16, 256 x 16B chunks) ----
#pragma unroll
        for (int p = 0; p < 2; ++p) {
            const int cidx = p * 128 + tid;    // 0..255
            const int r = cidx >> 3, part = (cidx & 7) * 8;
            async_copy16(c + (size_t)(kv0 + r) * ldc + 2 * D_MODEL + head * HEAD_DIM + part,
                         &sV[r * 136 + part]);
        }

        // ---- S = Q K^T (16x32 tile per wave) ----
        v8f s[2];
        s[0] = v8f_zero(); s[1] = v8f_zero();
#pragma unroll
        for (int ni = 0; ni < 2; ++ni) {
            const _Float16* krow =
                c + (size_t)(kv0 + ni * 16 + l15) * ldc + D_MODEL + head * HEAD_DIM;
#pragma unroll
            for (int kc = 0; kc < 4; ++kc) {
                v16h kf;
                v2h* kp = (v2h*)&kf;
#pragma unroll
                for (int jp = 0; jp < 8; ++jp) {
                    const int k = ((jp & 3) << 1) + kg + ((jp >> 2) << 4);
                    kp[jp] = *(const v2h*)(krow + kc * 32 + k);
                }
                s[ni] = __builtin_amdgcn_wmma_f32_16x16x32_f16(
                    false, qf[kc], false, kf, (short)0, s[ni], false, false);
            }
        }

        // ---- causal mask + online softmax (rows: half*8 + i) ----
        float alpha[8];
#pragma unroll
        for (int i = 0; i < 8; ++i) {
            const int q = qbase + half * 8 + i;
            float s0 = s[0][i] * scale;
            float s1 = s[1][i] * scale;
            if (kv0 + l15 > q)      s0 = -INFINITY;
            if (kv0 + 16 + l15 > q) s1 = -INFINITY;
            float mx = fmaxf(s0, s1);
#pragma unroll
            for (int off = 1; off < 16; off <<= 1)
                mx = fmaxf(mx, __shfl_xor(mx, off, 32));
            const float mnew = fmaxf(mrow[i], mx);
            const float a  = (mnew == -INFINITY) ? 1.0f : __expf(mrow[i] - mnew);
            const float p0 = (s0 == -INFINITY) ? 0.0f : __expf(s0 - mnew);
            const float p1 = (s1 == -INFINITY) ? 0.0f : __expf(s1 - mnew);
            float ssum = p0 + p1;
#pragma unroll
            for (int off = 1; off < 16; off <<= 1)
                ssum += __shfl_xor(ssum, off, 32);
            lrow[i]  = lrow[i] * a + ssum;
            mrow[i]  = mnew;
            alpha[i] = a;
            sP[wid][(half * 8 + i) * 32 + l15]      = (_Float16)p0;
            sP[wid][(half * 8 + i) * 32 + 16 + l15] = (_Float16)p1;
        }
        wait_async();
        __syncthreads();

        // ---- rescale O, then O += P V ----
#pragma unroll
        for (int vt = 0; vt < 8; ++vt)
#pragma unroll
            for (int i = 0; i < 8; ++i) o[vt][i] *= alpha[i];

        v16h pf;
        {
            v2h* pp = (v2h*)&pf;
#pragma unroll
            for (int jp = 0; jp < 8; ++jp) {
                const int k = ((jp & 3) << 1) + kg + ((jp >> 2) << 4);
                pp[jp] = *(const v2h*)&sP[wid][l15 * 32 + k];
            }
        }
#pragma unroll
        for (int vt = 0; vt < 8; ++vt) {
            v16h vf;
#pragma unroll
            for (int j = 0; j < 16; ++j) {
                const int kr = (j & 7) + kg + ((j >> 3) << 4);
                vf[j] = sV[kr * 136 + vt * 16 + l15];
            }
            o[vt] = __builtin_amdgcn_wmma_f32_16x16x32_f16(
                false, pf, false, vf, (short)0, o[vt], false, false);
        }
        __syncthreads();
    }

    // ---- normalize and store (f16 for the projection GEMM) ----
#pragma unroll
    for (int vt = 0; vt < 8; ++vt)
#pragma unroll
        for (int i = 0; i < 8; ++i) {
            const int q = qbase + half * 8 + i;
            out[(size_t)q * D_MODEL + head * HEAD_DIM + vt * 16 + l15] =
                (_Float16)(o[vt][i] / lrow[i]);
        }
}

// ---------------------------------------------------------------------------
// out[row] = LN(a[row]+b[row])*g + beta ; optional f16 copy for next GEMM.
// ---------------------------------------------------------------------------
__global__ __launch_bounds__(256) void add_layernorm(
    const float* __restrict__ a, const float* __restrict__ b,
    const float* __restrict__ g, const float* __restrict__ beta,
    float* __restrict__ out, _Float16* __restrict__ out_h)
{
    __shared__ float red[2][8];
    const int row = blockIdx.x;
    const int tid = threadIdx.x;
    const int lane = tid & 31, wid = tid >> 5;
    const float* pa = a + (size_t)row * D_MODEL;
    const float* pb = b + (size_t)row * D_MODEL;
    float x[8];
    float s = 0.0f, s2 = 0.0f;
#pragma unroll
    for (int e = 0; e < 8; ++e) {
        const int cidx = tid + e * 256;
        const float v = pa[cidx] + pb[cidx];
        x[e] = v; s += v; s2 += v * v;
    }
#pragma unroll
    for (int off = 1; off < 32; off <<= 1) {
        s  += __shfl_xor(s, off, 32);
        s2 += __shfl_xor(s2, off, 32);
    }
    if (lane == 0) { red[0][wid] = s; red[1][wid] = s2; }
    __syncthreads();
    float ts = 0.0f, ts2 = 0.0f;
#pragma unroll
    for (int w = 0; w < 8; ++w) { ts += red[0][w]; ts2 += red[1][w]; }
    const float inv_n = 1.0f / (float)D_MODEL;
    const float mu  = ts * inv_n;
    const float var = ts2 * inv_n - mu * mu;
    const float r   = rsqrtf(var + 1e-5f);
    float* po = out + (size_t)row * D_MODEL;
#pragma unroll
    for (int e = 0; e < 8; ++e) {
        const int cidx = tid + e * 256;
        const float v = (x[e] - mu) * r * g[cidx] + beta[cidx];
        po[cidx] = v;
        if (out_h != nullptr) out_h[(size_t)row * D_MODEL + cidx] = (_Float16)v;
    }
}

// ---------------------------------------------------------------------------
extern "C" void kernel_launch(void* const* d_in, const int* in_sizes, int n_in,
                              void* d_out, int out_size, void* d_ws, size_t ws_size,
                              hipStream_t stream)
{
    (void)in_sizes; (void)n_in; (void)out_size; (void)ws_size;

    const float* x     = (const float*)d_in[0];
    const float* C_w   = (const float*)d_in[1];
    const float* C_b   = (const float*)d_in[2];
    const float* lin_w = (const float*)d_in[3];
    const float* lin_b = (const float*)d_in[4];
    const float* ff1_w = (const float*)d_in[5];
    const float* ff1_b = (const float*)d_in[6];
    const float* ff2_w = (const float*)d_in[7];
    const float* ff2_b = (const float*)d_in[8];
    const float* ln1_g = (const float*)d_in[9];
    const float* ln1_b = (const float*)d_in[10];
    const float* ln2_g = (const float*)d_in[11];
    const float* ln2_b = (const float*)d_in[12];
    float* out = (float*)d_out;

    // ---- workspace layout (bytes) ----
    char* base = (char*)d_ws;
    size_t off = 0;
    auto alloc = [&](size_t bytes) { char* p = base + off; off += bytes; return p; };
    _Float16* C_wt   = (_Float16*)alloc((size_t)3 * D_MODEL * D_MODEL * 2); // [6144,2048]
    _Float16* lin_wt = (_Float16*)alloc((size_t)D_MODEL * D_MODEL * 2);     // [2048,2048]
    _Float16* ff1_wt = (_Float16*)alloc((size_t)INNER * D_MODEL * 2);       // [8192,2048]
    _Float16* ff2_wt = (_Float16*)alloc((size_t)D_MODEL * INNER * 2);       // [2048,8192]
    _Float16* x_h    = (_Float16*)alloc((size_t)SEQ * D_MODEL * 2);
    _Float16* c_h    = (_Float16*)alloc((size_t)SEQ * 3 * D_MODEL * 2);     // QKV (f16)
    _Float16* attn_h = (_Float16*)alloc((size_t)SEQ * D_MODEL * 2);
    float*    proj_f = (float*)alloc((size_t)SEQ * D_MODEL * 4);
    float*    h1_f   = (float*)alloc((size_t)SEQ * D_MODEL * 4);
    _Float16* h1_h   = (_Float16*)alloc((size_t)SEQ * D_MODEL * 2);
    _Float16* ffact_h = c_h;      // reuse: c_h+attn_h region (exactly SEQ*INNER f16)
    float*    ffout_f = proj_f;   // reuse: proj dead after LN1

    const dim3 blk256(256);
    const dim3 tblk(32, 8);

    // ---- 0) weight prep: f16 + transpose to [N,K]; x -> f16 ----
    transpose_f32_f16<<<dim3(3 * D_MODEL / 32, D_MODEL / 32), tblk, 0, stream>>>(
        C_w, C_wt, D_MODEL, 3 * D_MODEL);
    transpose_f32_f16<<<dim3(D_MODEL / 32, D_MODEL / 32), tblk, 0, stream>>>(
        lin_w, lin_wt, D_MODEL, D_MODEL);
    transpose_f32_f16<<<dim3(INNER / 32, D_MODEL / 32), tblk, 0, stream>>>(
        ff1_w, ff1_wt, D_MODEL, INNER);
    transpose_f32_f16<<<dim3(D_MODEL / 32, INNER / 32), tblk, 0, stream>>>(
        ff2_w, ff2_wt, INNER, D_MODEL);
    cvt_f32_f16<<<(SEQ * D_MODEL / 4 + 255) / 256, blk256, 0, stream>>>(
        x, x_h, SEQ * D_MODEL / 4);

    // ---- 1) QKV: c_h = f16(x @ C_w + C_b)          [2048 x 6144] ----
    gemm_wmma_f16<0, false, true><<<dim3(3 * D_MODEL / BN, SEQ / BM), blk256, 0, stream>>>(
        x_h, C_wt, C_b, nullptr, c_h, 3 * D_MODEL, D_MODEL);

    // ---- 2) causal flash attention -> attn_h (f16) ----
    attn_flash<<<dim3(HEADS, SEQ / 64), dim3(128), 0, stream>>>(c_h, attn_h);

    // ---- 3) projection: proj_f = attn @ lin_w + b  [2048 x 2048] ----
    gemm_wmma_f16<0, true, false><<<dim3(D_MODEL / BN, SEQ / BM), blk256, 0, stream>>>(
        attn_h, lin_wt, lin_b, proj_f, nullptr, D_MODEL, D_MODEL);

    // ---- 4) h1 = LN(x + proj)  (f32 + f16 copy) ----
    add_layernorm<<<SEQ, blk256, 0, stream>>>(x, proj_f, ln1_g, ln1_b, h1_f, h1_h);

    // ---- 5) ffact = f16(GELU(h1 @ ff1_w + b))      [2048 x 8192] ----
    gemm_wmma_f16<1, false, true><<<dim3(INNER / BN, SEQ / BM), blk256, 0, stream>>>(
        h1_h, ff1_wt, ff1_b, nullptr, ffact_h, INNER, D_MODEL);

    // ---- 6) ffout = ffact @ ff2_w + b              [2048 x 2048] ----
    gemm_wmma_f16<0, true, false><<<dim3(D_MODEL / BN, SEQ / BM), blk256, 0, stream>>>(
        ffact_h, ff2_wt, ff2_b, ffout_f, nullptr, D_MODEL, INNER);

    // ---- 7) out = LN(h1 + ffout) ----
    add_layernorm<<<SEQ, blk256, 0, stream>>>(h1_f, ffout_f, ln2_g, ln2_b, out, nullptr);
}
